// GIN_70686571758165
// MI455X (gfx1250) — matrix operand order repaired and checked by reading, not compiled, not run on hardware
//
#include <hip/hip_runtime.h>

#define IN_DIM   4096
#define OUT_DIM  2048
#define NNODES   8192
#define KSAMP    5
#define BN_EPS   1e-5f

typedef __attribute__((ext_vector_type(16))) __bf16       v16bf;
typedef __attribute__((ext_vector_type(8)))  float        v8f;
typedef __attribute__((ext_vector_type(4)))  float        v4f;
typedef __attribute__((ext_vector_type(4)))  __bf16       v4bf;
typedef __attribute__((ext_vector_type(4)))  unsigned int v4u;
typedef __attribute__((ext_vector_type(4)))  int          v4i;

union Frag16 { v16bf v; v4u q[2]; };

// ---- CDNA5 async global->LDS path (guarded: falls back to reg staging) ----
#if defined(__gfx1250__) && __has_builtin(__builtin_amdgcn_global_load_async_to_lds_b128)
#define HAVE_ASYNC_LDS 1
typedef __attribute__((address_space(1))) v4i* gv4i_p;   // pointer to AS(1) v4i
typedef __attribute__((address_space(3))) v4i* lv4i_p;   // pointer to AS(3) v4i

__device__ __forceinline__ void async_copy_b128(const void* g, void* l) {
  __builtin_amdgcn_global_load_async_to_lds_b128(
      (gv4i_p)(void*)(g), (lv4i_p)(l), 0, 0);
}
#else
#define HAVE_ASYNC_LDS 0
#endif

#if __has_builtin(__builtin_amdgcn_s_wait_asynccnt)
#define ASYNC_WAIT0() __builtin_amdgcn_s_wait_asynccnt(0)
#else
#define ASYNC_WAIT0() asm volatile("s_wait_asynccnt 0x0" ::: "memory")
#endif

// ---------------------------------------------------------------------------
// Gather + GIN aggregate: H0[n][c] = bf16( 2*Wemb[nodes[n]][c] + sum_k Wemb[nbr[n][k]][c] )
// grid: (IN_DIM/1024, NNODES), block 256, 4 floats per thread
// ---------------------------------------------------------------------------
__global__ __launch_bounds__(256) void k_embed_agg(
    const int* __restrict__ nodes, const int* __restrict__ nbr,
    const float* __restrict__ Wemb, __bf16* __restrict__ H0)
{
  const int n = blockIdx.y;
  const int c = blockIdx.x * 1024 + threadIdx.x * 4;
  const int self = nodes[n];
  v4f acc = *(const v4f*)(Wemb + (size_t)self * IN_DIM + c);
  acc += acc;  // h = self_emb + (self_emb + sum_nb)
#pragma unroll
  for (int k = 0; k < KSAMP; ++k) {
    const int id = nbr[n * KSAMP + k];
    acc += *(const v4f*)(Wemb + (size_t)id * IN_DIM + c);
  }
  v4bf o;
  o.x = (__bf16)acc.x; o.y = (__bf16)acc.y;
  o.z = (__bf16)acc.z; o.w = (__bf16)acc.w;
  *(v4bf*)(H0 + (size_t)n * IN_DIM + c) = o;
}

// ---------------------------------------------------------------------------
// Weight prep: W[K][Nout] f32 -> WT[Nout][K] bf16 (so B-frags are K-contiguous)
// grid: (Nout/32, K/32), block 256 (32x8)
// ---------------------------------------------------------------------------
__global__ __launch_bounds__(256) void k_w_transpose_bf16(
    const float* __restrict__ W, __bf16* __restrict__ WT, int K, int Nout)
{
  __shared__ float tile[32][33];
  const int tx = threadIdx.x & 31;
  const int ty = threadIdx.x >> 5;
  const int k0 = blockIdx.y * 32;
  const int j0 = blockIdx.x * 32;
#pragma unroll
  for (int i = 0; i < 4; ++i) {
    const int k = ty + i * 8;
    tile[k][tx] = W[(size_t)(k0 + k) * Nout + j0 + tx];
  }
  __syncthreads();
#pragma unroll
  for (int i = 0; i < 4; ++i) {
    const int j = ty + i * 8;
    WT[(size_t)(j0 + j) * K + k0 + tx] = (__bf16)tile[tx][j];
  }
}

// ---------------------------------------------------------------------------
// GEMM: Z[M][Nout] f32 = A[M][K](bf16) x BT[Nout][K](bf16)^T + bias
// v_wmma_f32_16x16x32_bf16, block tile 128x128x32, 8 waves, wave tile 64x32.
// Double-buffered LDS (padded stride 48 bf16 = 96B). Tile feed uses
// GLOBAL_LOAD_ASYNC_TO_LDS_B128 (ASYNCcnt) when available.
// ---------------------------------------------------------------------------
#define BM 128
#define BN 128
#define BK 32
#define LDSS 48

__global__ __launch_bounds__(256) void k_gemm_bf16_wmma(
    const __bf16* __restrict__ A,   // [M][K]
    const __bf16* __restrict__ BT,  // [Nout][K]
    const float*  __restrict__ bias,
    float* __restrict__ Z,          // [M][Nout]
    int M, int Nout, int K)
{
  __shared__ __bf16 sA[2][BM * LDSS];   // 24 KB
  __shared__ __bf16 sB[2][BN * LDSS];   // 24 KB

  const int tid  = threadIdx.x;
  const int lane = tid & 31;
  const int wave = tid >> 5;
  const int wm   = (wave >> 2) * 64;    // 0 / 64
  const int wn   = (wave & 3) * 32;     // 0 / 32 / 64 / 96
  const int l15  = lane & 15;
  const int kg   = lane >> 4;           // K-group per ISA wave32 layout

  const int bm = blockIdx.y * BM;
  const int bn = blockIdx.x * BN;

  // tile feeders: 256 threads * 2 chunks of 16B each for A and B
  const int row_ld = tid >> 2;          // 0..63
  const int kof_ld = (tid & 3) * 8;     // 0,8,16,24 (bf16 elems)
  const __bf16* gA = A  + (size_t)(bm + row_ld) * K + kof_ld;
  const __bf16* gB = BT + (size_t)(bn + row_ld) * K + kof_ld;
  const size_t gstep = (size_t)64 * K;

  v8f acc[4][2];
  const v8f vzero = {0.f, 0.f, 0.f, 0.f, 0.f, 0.f, 0.f, 0.f};
#pragma unroll
  for (int fm = 0; fm < 4; ++fm)
#pragma unroll
    for (int fn = 0; fn < 2; ++fn) acc[fm][fn] = vzero;

  const int ntiles = K / BK;

#if HAVE_ASYNC_LDS
  // ---- async tile feed: global -> LDS without VGPR round trip ----
  {
    async_copy_b128(gA,         &sA[0][row_ld * LDSS + kof_ld]);
    async_copy_b128(gA + gstep, &sA[0][(row_ld + 64) * LDSS + kof_ld]);
    async_copy_b128(gB,         &sB[0][row_ld * LDSS + kof_ld]);
    async_copy_b128(gB + gstep, &sB[0][(row_ld + 64) * LDSS + kof_ld]);
  }
  ASYNC_WAIT0();
  __syncthreads();

  for (int kt = 0; kt < ntiles; ++kt) {
    const int cur = kt & 1;
    if (kt + 1 < ntiles) {      // kick next tile into the other LDS buffer
      const int nb = cur ^ 1;
      const int k0 = (kt + 1) * BK;
      async_copy_b128(gA + k0,         &sA[nb][row_ld * LDSS + kof_ld]);
      async_copy_b128(gA + gstep + k0, &sA[nb][(row_ld + 64) * LDSS + kof_ld]);
      async_copy_b128(gB + k0,         &sB[nb][row_ld * LDSS + kof_ld]);
      async_copy_b128(gB + gstep + k0, &sB[nb][(row_ld + 64) * LDSS + kof_ld]);
    }

    Frag16 af[4];
#pragma unroll
    for (int fm = 0; fm < 4; ++fm) {
      const __bf16* p = &sA[cur][(wm + fm * 16 + l15) * LDSS];
      af[fm].q[0] = *(const v4u*)(p + 8 * kg);
      af[fm].q[1] = *(const v4u*)(p + 16 + 8 * kg);
    }
    Frag16 bfr[2];
#pragma unroll
    for (int fn = 0; fn < 2; ++fn) {
      const __bf16* p = &sB[cur][(wn + fn * 16 + l15) * LDSS + 8 * kg];
      bfr[fn].q[0] = *(const v4u*)(p);
      bfr[fn].q[1] = *(const v4u*)(p + 16);
    }

#pragma unroll
    for (int fm = 0; fm < 4; ++fm)
#pragma unroll
      for (int fn = 0; fn < 2; ++fn)
        acc[fm][fn] = __builtin_amdgcn_wmma_f32_16x16x32_bf16(
            false, af[fm].v, false, bfr[fn].v, (short)0, acc[fm][fn],
            false, false);

    ASYNC_WAIT0();      // next tile fully landed in LDS
    __syncthreads();    // everyone done reading cur + sees new tile
  }
#else
  // ---- fallback: register-staged feed ----
  {
    const v4u a0 = *(const v4u*)(gA);
    const v4u a1 = *(const v4u*)(gA + gstep);
    const v4u b0 = *(const v4u*)(gB);
    const v4u b1 = *(const v4u*)(gB + gstep);
    *(v4u*)&sA[0][row_ld * LDSS + kof_ld]        = a0;
    *(v4u*)&sA[0][(row_ld + 64) * LDSS + kof_ld] = a1;
    *(v4u*)&sB[0][row_ld * LDSS + kof_ld]        = b0;
    *(v4u*)&sB[0][(row_ld + 64) * LDSS + kof_ld] = b1;
  }
  __syncthreads();

  for (int kt = 0; kt < ntiles; ++kt) {
    const int cur = kt & 1;
    v4u a0, a1, b0, b1;
    const bool more = (kt + 1) < ntiles;
    if (more) {
      const int k0 = (kt + 1) * BK;
      a0 = *(const v4u*)(gA + k0);
      a1 = *(const v4u*)(gA + gstep + k0);
      b0 = *(const v4u*)(gB + k0);
      b1 = *(const v4u*)(gB + gstep + k0);
    }

    Frag16 af[4];
#pragma unroll
    for (int fm = 0; fm < 4; ++fm) {
      const __bf16* p = &sA[cur][(wm + fm * 16 + l15) * LDSS];
      af[fm].q[0] = *(const v4u*)(p + 8 * kg);
      af[fm].q[1] = *(const v4u*)(p + 16 + 8 * kg);
    }
    Frag16 bfr[2];
#pragma unroll
    for (int fn = 0; fn < 2; ++fn) {
      const __bf16* p = &sB[cur][(wn + fn * 16 + l15) * LDSS + 8 * kg];
      bfr[fn].q[0] = *(const v4u*)(p);
      bfr[fn].q[1] = *(const v4u*)(p + 16);
    }

#pragma unroll
    for (int fm = 0; fm < 4; ++fm)
#pragma unroll
      for (int fn = 0; fn < 2; ++fn)
        acc[fm][fn] = __builtin_amdgcn_wmma_f32_16x16x32_bf16(
            false, af[fm].v, false, bfr[fn].v, (short)0, acc[fm][fn],
            false, false);

    __syncthreads();
    if (more) {
      const int nb = cur ^ 1;
      *(v4u*)&sA[nb][row_ld * LDSS + kof_ld]        = a0;
      *(v4u*)&sA[nb][(row_ld + 64) * LDSS + kof_ld] = a1;
      *(v4u*)&sB[nb][row_ld * LDSS + kof_ld]        = b0;
      *(v4u*)&sB[nb][(row_ld + 64) * LDSS + kof_ld] = b1;
    }
    __syncthreads();
  }
#endif

  // epilogue: D layout -> lane l15 is column, VGPR r is row M = 8*kg + r
#pragma unroll
  for (int fm = 0; fm < 4; ++fm) {
    const int row = bm + wm + fm * 16 + kg * 8;
#pragma unroll
    for (int fn = 0; fn < 2; ++fn) {
      const int col = bn + wn + fn * 16 + l15;
      const float bb = bias[col];
      float* zp = Z + (size_t)row * Nout + col;
#pragma unroll
      for (int r = 0; r < 8; ++r)
        zp[(size_t)r * Nout] = acc[fm][fn][r] + bb;
    }
  }
}

// ---------------------------------------------------------------------------
// Training-mode BN stats per column (biased var) -> fused scale/shift
// grid: Nout/64, block 256 (64 cols x 4 row-groups)
// ---------------------------------------------------------------------------
__global__ __launch_bounds__(256) void k_colstats(
    const float* __restrict__ Z, const float* __restrict__ g,
    const float* __restrict__ be, float* __restrict__ scale,
    float* __restrict__ shift, int M, int Nout)
{
  __shared__ float ssum[256], ssq[256];
  const int col = threadIdx.x & 63;
  const int rg  = threadIdx.x >> 6;
  const int j = blockIdx.x * 64 + col;
  float sum = 0.f, sq = 0.f;
  for (int r = rg; r < M; r += 4) {
    const float v = Z[(size_t)r * Nout + j];
    sum += v; sq += v * v;
  }
  ssum[threadIdx.x] = sum; ssq[threadIdx.x] = sq;
  __syncthreads();
  if (rg == 0) {
#pragma unroll
    for (int i = 1; i < 4; ++i) { sum += ssum[col + 64 * i]; sq += ssq[col + 64 * i]; }
    const float inv  = 1.f / (float)M;
    const float mean = sum * inv;
    const float var  = sq * inv - mean * mean;
    const float sc   = g[j] * rsqrtf(var + BN_EPS);
    scale[j] = sc;
    shift[j] = be[j] - mean * sc;
  }
}

// ---------------------------------------------------------------------------
// apply BN + ReLU, emit bf16 activation matrix for the next GEMM
// ---------------------------------------------------------------------------
__global__ __launch_bounds__(256) void k_bn_relu_to_bf16(
    const float* __restrict__ Z, const float* __restrict__ scale,
    const float* __restrict__ shift, __bf16* __restrict__ Aout)
{
  const size_t i = ((size_t)blockIdx.x * 256 + threadIdx.x) * 4;
  const int j = (int)(i & (size_t)(OUT_DIM - 1));
  const v4f z = *(const v4f*)(Z + i);
  const v4f s = *(const v4f*)(scale + j);
  const v4f t = *(const v4f*)(shift + j);
  v4bf o;
  o.x = (__bf16)fmaxf(z.x * s.x + t.x, 0.f);
  o.y = (__bf16)fmaxf(z.y * s.y + t.y, 0.f);
  o.z = (__bf16)fmaxf(z.z * s.z + t.z, 0.f);
  o.w = (__bf16)fmaxf(z.w * s.w + t.w, 0.f);
  *(v4bf*)(Aout + i) = o;
}

// ---------------------------------------------------------------------------
// final BN + ReLU fused with transpose: out[j][n] (f32, [OUT][N])
// grid: (Nout/32, M/32), block 256
// ---------------------------------------------------------------------------
__global__ __launch_bounds__(256) void k_bn_relu_transpose_out(
    const float* __restrict__ Z, const float* __restrict__ scale,
    const float* __restrict__ shift, float* __restrict__ out, int M, int Nout)
{
  __shared__ float tile[32][33];
  const int tx = threadIdx.x & 31;
  const int ty = threadIdx.x >> 5;
  const int n0 = blockIdx.y * 32;
  const int j0 = blockIdx.x * 32;
  const float sc = scale[j0 + tx];
  const float sh = shift[j0 + tx];
#pragma unroll
  for (int i = 0; i < 4; ++i) {
    const int n = ty + i * 8;
    const float v = Z[(size_t)(n0 + n) * Nout + j0 + tx];
    tile[n][tx] = fmaxf(v * sc + sh, 0.f);
  }
  __syncthreads();
#pragma unroll
  for (int i = 0; i < 4; ++i) {
    const int j = ty + i * 8;
    out[(size_t)(j0 + j) * M + n0 + tx] = tile[tx][j];
  }
}

// ---------------------------------------------------------------------------
extern "C" void kernel_launch(void* const* d_in, const int* in_sizes, int n_in,
                              void* d_out, int out_size, void* d_ws, size_t ws_size,
                              hipStream_t stream)
{
  const int*   nodes = (const int*)d_in[0];
  const int*   nbr   = (const int*)d_in[1];
  const float* Wemb  = (const float*)d_in[2];
  const float* W0a   = (const float*)d_in[3];
  const float* b0a   = (const float*)d_in[4];
  const float* g0a   = (const float*)d_in[5];
  const float* be0a  = (const float*)d_in[6];
  const float* W0b   = (const float*)d_in[7];
  const float* b0b   = (const float*)d_in[8];
  const float* gbn0  = (const float*)d_in[9];
  const float* bbn0  = (const float*)d_in[10];
  const float* W1a   = (const float*)d_in[11];
  const float* b1a   = (const float*)d_in[12];
  const float* g1a   = (const float*)d_in[13];
  const float* be1a  = (const float*)d_in[14];
  const float* W1b   = (const float*)d_in[15];
  const float* b1b   = (const float*)d_in[16];
  const float* gbn1  = (const float*)d_in[17];
  const float* bbn1  = (const float*)d_in[18];
  float* out = (float*)d_out;
  (void)in_sizes; (void)n_in; (void)out_size; (void)ws_size;

  // workspace layout (~200 MiB)
  char* p = (char*)d_ws;
  size_t off = 0;
  auto take = [&](size_t bytes) {
    void* r = p + off;
    off = (off + bytes + 255) & ~(size_t)255;
    return r;
  };
  __bf16* H0   = (__bf16*)take((size_t)NNODES * IN_DIM * 2);
  __bf16* Ab   = (__bf16*)take((size_t)NNODES * OUT_DIM * 2);
  float*  Zb   = (float*) take((size_t)NNODES * OUT_DIM * 4);
  __bf16* WT0a = (__bf16*)take((size_t)IN_DIM  * OUT_DIM * 2);
  __bf16* WT0b = (__bf16*)take((size_t)OUT_DIM * OUT_DIM * 2);
  __bf16* WT1a = (__bf16*)take((size_t)OUT_DIM * OUT_DIM * 2);
  __bf16* WT1b = (__bf16*)take((size_t)OUT_DIM * OUT_DIM * 2);
  float*  sc   = (float*) take((size_t)OUT_DIM * 4);
  float*  sh   = (float*) take((size_t)OUT_DIM * 4);

  const dim3 blk(256);
  const dim3 ggrid(OUT_DIM / BN, NNODES / BM);          // (16, 64)
  const dim3 sgrid(OUT_DIM / 64);                       // 32
  const dim3 agrid((unsigned)((size_t)NNODES * OUT_DIM / 1024));  // 16384
  const dim3 tgrid(OUT_DIM / 32, NNODES / 32);          // (64, 256)

  // one-time weight transpose+convert
  k_w_transpose_bf16<<<dim3(OUT_DIM/32, IN_DIM/32),  blk, 0, stream>>>(W0a, WT0a, IN_DIM,  OUT_DIM);
  k_w_transpose_bf16<<<dim3(OUT_DIM/32, OUT_DIM/32), blk, 0, stream>>>(W0b, WT0b, OUT_DIM, OUT_DIM);
  k_w_transpose_bf16<<<dim3(OUT_DIM/32, OUT_DIM/32), blk, 0, stream>>>(W1a, WT1a, OUT_DIM, OUT_DIM);
  k_w_transpose_bf16<<<dim3(OUT_DIM/32, OUT_DIM/32), blk, 0, stream>>>(W1b, WT1b, OUT_DIM, OUT_DIM);

  // gather + aggregate (W_emb is 64MB -> resident in 192MB L2)
  k_embed_agg<<<dim3(IN_DIM/1024, NNODES), blk, 0, stream>>>(nodes, nbr, Wemb, H0);

  // layer 0
  k_gemm_bf16_wmma<<<ggrid, blk, 0, stream>>>(H0, WT0a, b0a, Zb, NNODES, OUT_DIM, IN_DIM);
  k_colstats<<<sgrid, blk, 0, stream>>>(Zb, g0a, be0a, sc, sh, NNODES, OUT_DIM);
  k_bn_relu_to_bf16<<<agrid, blk, 0, stream>>>(Zb, sc, sh, Ab);
  k_gemm_bf16_wmma<<<ggrid, blk, 0, stream>>>(Ab, WT0b, b0b, Zb, NNODES, OUT_DIM, OUT_DIM);
  k_colstats<<<sgrid, blk, 0, stream>>>(Zb, gbn0, bbn0, sc, sh, NNODES, OUT_DIM);
  k_bn_relu_to_bf16<<<agrid, blk, 0, stream>>>(Zb, sc, sh, Ab);

  // layer 1
  k_gemm_bf16_wmma<<<ggrid, blk, 0, stream>>>(Ab, WT1a, b1a, Zb, NNODES, OUT_DIM, OUT_DIM);
  k_colstats<<<sgrid, blk, 0, stream>>>(Zb, g1a, be1a, sc, sh, NNODES, OUT_DIM);
  k_bn_relu_to_bf16<<<agrid, blk, 0, stream>>>(Zb, sc, sh, Ab);
  k_gemm_bf16_wmma<<<ggrid, blk, 0, stream>>>(Ab, WT1b, b1b, Zb, NNODES, OUT_DIM, OUT_DIM);
  k_colstats<<<sgrid, blk, 0, stream>>>(Zb, gbn1, bbn1, sc, sh, NNODES, OUT_DIM);

  // final BN+ReLU fused with transpose to [OUT][N]
  k_bn_relu_transpose_out<<<tgrid, blk, 0, stream>>>(Zb, sc, sh, out, NNODES, OUT_DIM);
}
